// SAMBlock_20151986553014
// MI455X (gfx1250) — matrix operand order repaired and checked by reading, not compiled, not run on hardware
//
#include <hip/hip_runtime.h>

// ---- types -----------------------------------------------------------------
typedef __attribute__((ext_vector_type(16))) __bf16      v16bf;
typedef __attribute__((ext_vector_type(8)))  float       v8f;
typedef __attribute__((ext_vector_type(4)))  unsigned    u32x4;
typedef __attribute__((ext_vector_type(2)))  unsigned    u32x2;
typedef __attribute__((ext_vector_type(4)))  float       f32x4;

#define NFIELDS 24
#define DPAD    32      // n padded to 32 (two 16-row tiles); pad rows are zero
#define D       64
#define NUM_LAYERS 2
#define WPB     4       // waves per block (one batch per wave)

// ---- bf16 conversion: native packed cvt (RNE) ------------------------------
__device__ __forceinline__ unsigned pk_bf(float a, float b) {
    unsigned r;
    asm("v_cvt_pk_bf16_f32 %0, %1, %2" : "=v"(r) : "v"(a), "v"(b));
    return r;
}
__device__ __forceinline__ unsigned short f2bf1(float x) {
    return (unsigned short)pk_bf(x, x);
}
__device__ __forceinline__ float bf2f(unsigned short s) {
    unsigned u = (unsigned)s << 16;
    return __builtin_bit_cast(float, u);
}

union FragU { v16bf v; u32x4 q[2]; };

// A fragment, 16-bit 16x32 tile (ISA 7.12.2): lane<16 -> M=lane, K={0..7,16..23};
// lane>=16 -> M=lane-16, K={8..15,24..31}. X is row-major [rows][ld] bf16.
__device__ __forceinline__ v16bf loadA(const unsigned short* X, int ld,
                                       int r0, int k0, int lo, int hi) {
    FragU f;
    const unsigned short* p = X + (r0 + lo) * ld + k0 + hi * 8;
    f.q[0] = *(const u32x4*)(p);        // K = k0 + hi*8 .. +7
    f.q[1] = *(const u32x4*)(p + 16);   // K = k0 + 16 + hi*8 .. +7
    return f.v;
}

// B fragment, 16-bit 32x16 tile: lane holds column N = n0+lane%16, K = k0+{0..7,16..23} (+8 for lane>=16).
__device__ __forceinline__ v16bf loadB(const unsigned short* X, int ld,
                                       int k0, int n0, int lane) {
    union { v16bf v; unsigned short u[16]; } f;
    const unsigned short* p = X + (k0 + (lane >> 4) * 8) * ld + n0 + (lane & 15);
#pragma unroll
    for (int s = 0; s < 8; ++s) {
        f.u[s] = p[s * ld];
        f.u[8 + s] = p[(16 + s) * ld];
    }
    return f.v;
}

__device__ __forceinline__ v8f wmma_bf16(v16bf a, v16bf b, v8f c) {
    return __builtin_amdgcn_wmma_f32_16x16x32_bf16(
        /*neg_a=*/false, a, /*neg_b=*/false, b,
        /*c_mod=*/(short)0, c, /*reuse_a=*/false, /*reuse_b=*/false);
}

// ---- prologue: transpose + bf16-convert the 4 weight matrices into d_ws ----
// Wt layout (bf16): [0..4095]=Wk^T L0, [4096..]=Wk^T L1, [8192..]=Wq^T L0, [12288..]=Wq^T L1
__global__ void prep_weights(const float* __restrict__ Kw,
                             const float* __restrict__ Qw,
                             unsigned short* __restrict__ Wt) {
    int t = threadIdx.x;                 // 256 threads, 16384 elems -> 64 each
    for (int i = 0; i < 64; ++i) {
        int idx = t * 64 + i;
        int mat = idx >> 12;             // 0,1 = K layers; 2,3 = Q layers
        int r   = idx & 4095;            // transposed position: r = d*64 + e
        int d = r >> 6, e = r & 63;
        const float* W = (mat < 2) ? (Kw + mat * 4096) : (Qw + (mat - 2) * 4096);
        Wt[idx] = f2bf1(W[e * 64 + d]);  // Wt[d][e] = W[e][d]  (x @ W^T)
    }
}

// ---- main fused kernel: one wave per batch, both layers in LDS -------------
__global__ void __launch_bounds__(32 * WPB)
sam3e_kernel(const float* __restrict__ Fin,
             const unsigned short* __restrict__ Wt,
             float* __restrict__ Out, int nB) {
    __shared__ __align__(16) unsigned short bbuf[WPB][2][DPAD * D];  // bf16 F ping-pong
    __shared__ __align__(16) unsigned short kft [WPB][D * DPAD];     // KF^T d-major
    __shared__ __align__(16) unsigned short sbf [WPB][DPAD * DPAD];  // S bf16

    const int wave = threadIdx.x >> 5;
    const int lane = threadIdx.x & 31;
    const int lo = lane & 15, hi = lane >> 4;
    const int batch = blockIdx.x * WPB + wave;
    if (batch >= nB) return;

    unsigned short* cbf = bbuf[wave][0];
    unsigned short* nbf = bbuf[wave][1];
    unsigned short* kf  = kft[wave];
    unsigned short* sb  = sbf[wave];

    // load F [24][64] -> bf16 rows 0..23, zero pad rows 24..31
    const float* src = Fin + (size_t)batch * (NFIELDS * D);
#pragma unroll
    for (int i = 0; i < 12; ++i) {
        int fl = (i * 32 + lane) * 4;
        f32x4 x = *(const f32x4*)(src + fl);
        u32x2 p;
        p[0] = pk_bf(x[0], x[1]);
        p[1] = pk_bf(x[2], x[3]);
        *(u32x2*)(cbf + fl) = p;
    }
    u32x2 z2 = {0u, 0u};
#pragma unroll
    for (int i = 0; i < 4; ++i) {
        int fl = (i * 32 + lane) * 4;
        *(u32x2*)(cbf + NFIELDS * D + fl) = z2;
    }

    float* dst = Out + (size_t)batch * (NFIELDS * D);

#pragma unroll
    for (int l = 0; l < NUM_LAYERS; ++l) {
        const unsigned short* WkT = Wt + l * 4096;
        const unsigned short* WqT = Wt + 2 * 4096 + l * 4096;
        const bool last = (l == NUM_LAYERS - 1);

        // KF = F @ Wk^T, stored transposed: kf[d][m]
        // (consecutive acc elements = consecutive m -> one b128 store)
        for (int r = 0; r < DPAD; r += 16)
            for (int e0 = 0; e0 < D; e0 += 16) {
                v8f acc = {};
#pragma unroll
                for (int k0 = 0; k0 < D; k0 += 32)
                    acc = wmma_bf16(loadA(cbf, D, r, k0, lo, hi),
                                    loadB(WkT, D, k0, e0, lane), acc);
                u32x4 pk;
                pk[0] = pk_bf(acc[0], acc[1]);
                pk[1] = pk_bf(acc[2], acc[3]);
                pk[2] = pk_bf(acc[4], acc[5]);
                pk[3] = pk_bf(acc[6], acc[7]);
                *(u32x4*)(kf + (e0 + lo) * DPAD + r + 8 * hi) = pk;
            }

        // S = F @ KF^T   (B[k=d][n=m] = kf[d][m], contiguous)
        for (int r = 0; r < DPAD; r += 16)
            for (int n0 = 0; n0 < DPAD; n0 += 16) {
                v8f acc = {};
#pragma unroll
                for (int k0 = 0; k0 < D; k0 += 32)
                    acc = wmma_bf16(loadA(cbf, D, r, k0, lo, hi),
                                    loadB(kf, DPAD, k0, n0, lane), acc);
#pragma unroll
                for (int v = 0; v < 8; ++v)
                    sb[(r + v + 8 * hi) * DPAD + (n0 + lo)] = f2bf1(acc[v]);
            }

        // out = F * (S @ F) + F @ Wq^T   (fused epilogue per 16x16 tile)
        // non-last layer: write bf16 next-F to LDS; last layer: fp32 straight to HBM
        for (int r = 0; r < DPAD; r += 16)
            for (int n0 = 0; n0 < D; n0 += 16) {
                v8f t = {};
                t = wmma_bf16(loadA(sb, DPAD, r, 0, lo, hi),   // K = 32 (fields)
                              loadB(cbf, D, 0, n0, lane), t);
                v8f q = {};
#pragma unroll
                for (int k0 = 0; k0 < D; k0 += 32)
                    q = wmma_bf16(loadA(cbf, D, r, k0, lo, hi),
                                  loadB(WqT, D, k0, n0, lane), q);
                if (!last) {
#pragma unroll
                    for (int v = 0; v < 8; ++v) {
                        int m = r + v + 8 * hi, n = n0 + lo;
                        float o = bf2f(cbf[m * D + n]) * t[v] + q[v];
                        nbf[m * D + n] = f2bf1(o);
                    }
                } else {
#pragma unroll
                    for (int v = 0; v < 8; ++v) {
                        int m = r + v + 8 * hi, n = n0 + lo;
                        if (m < NFIELDS)   // uniform per half-wave; skips pad rows
                            dst[m * D + n] = bf2f(cbf[m * D + n]) * t[v] + q[v];
                    }
                }
            }

        unsigned short* tb = cbf; cbf = nbf; nbf = tb;
    }
}

extern "C" void kernel_launch(void* const* d_in, const int* in_sizes, int n_in,
                              void* d_out, int out_size, void* d_ws, size_t ws_size,
                              hipStream_t stream) {
    const float* F  = (const float*)d_in[0];
    const float* Kw = (const float*)d_in[1];
    const float* Qw = (const float*)d_in[2];
    unsigned short* Wt = (unsigned short*)d_ws;   // 16384 bf16 = 32 KB scratch
    int nB = in_sizes[0] / (NFIELDS * D);         // 8192

    prep_weights<<<1, 256, 0, stream>>>(Kw, Qw, Wt);
    int blocks = (nB + WPB - 1) / WPB;            // 4 waves/block, 1 batch/wave
    sam3e_kernel<<<blocks, 32 * WPB, 0, stream>>>(F, Wt, (float*)d_out, nB);
}
